// MultiHeadAttention_15427522527750
// MI455X (gfx1250) — compile-verified, matrix operand order checked by reading, not run on hardware
//
#include <hip/hip_runtime.h>
#include <hip/hip_bf16.h>

typedef __attribute__((ext_vector_type(16))) __bf16 v16bf;
typedef __attribute__((ext_vector_type(8)))  __bf16 v8bf;
typedef __attribute__((ext_vector_type(8)))  float  v8f;

#define BB 4
#define LL 2048
#define DD 1024
#define HH 16
#define DH 64
#define SCALE 0.125f   // DH^-0.5

__device__ __forceinline__ v8f wmma_bf16(v16bf a, v16bf b, v8f c) {
  return __builtin_amdgcn_wmma_f32_16x16x32_bf16(false, a, false, b, (short)0, c, false, false);
}

// Load one 16x32 bf16 operand tile (A-layout; B-layout is identical with
// "row" = output column since W is stored [n,k]).
// lane L: row = base + (L&15); khalf = L>>4;
//   VGPR0-3 <- K = kbase + khalf*8 + 0..7 ; VGPR4-7 <- K = kbase + 16 + khalf*8 + 0..7
__device__ __forceinline__ v16bf load_tile16x32(const __bf16* __restrict__ src,
                                                int rowbase, int ld, int kbase, int lane) {
  const int r  = rowbase + (lane & 15);
  const int kh = (lane >> 4) * 8;
  const __bf16* p = src + (long)r * ld + kbase + kh;
  union { v16bf v; v8bf h[2]; } u;
  u.h[0] = *(const v8bf*)(p);
  u.h[1] = *(const v8bf*)(p + 16);
  return u.v;
}

// fp32 -> bf16, 8 elements/thread (b128 in, b128 out)
__global__ void cvt_f32_bf16_x8(const float4* __restrict__ in, v8bf* __restrict__ out, int n8) {
  int i = blockIdx.x * blockDim.x + threadIdx.x;
  if (i < n8) {
    const float4 f0 = in[2 * i];
    const float4 f1 = in[2 * i + 1];
    v8bf o;
    o[0] = (__bf16)f0.x; o[1] = (__bf16)f0.y; o[2] = (__bf16)f0.z; o[3] = (__bf16)f0.w;
    o[4] = (__bf16)f1.x; o[5] = (__bf16)f1.y; o[6] = (__bf16)f1.z; o[7] = (__bf16)f1.w;
    out[i] = o;
  }
}

__global__ void mask_bias(const int* __restrict__ m, float* __restrict__ bias, int n) {
  int i = blockIdx.x * blockDim.x + threadIdx.x;
  if (i < n) bias[i] = (m[i] == 1) ? 0.f : -1e9f;
}

// C[m,n] = sum_k A[m,k] * W[n,k];  M=8192, N=K=1024.
// Block = 4 waves computing a 256x64 output region (wave w owns rows
// [tmb*256 + w*64, +64), all waves share one 64-wide N block).
// Per K-step the block stages the 64x32 bf16 weight panel (4 KB) into LDS
// once via global_load_async_to_lds_b128 (ASYNCcnt), then each wave reads
// its B tiles with ds_load_b128 and issues 16 v_wmma against 4 direct
// global b128 A-tile loads.
// MODE 0: bf16 out, head-split [B,H,L,DH]
// MODE 2: bf16 out, transposed heads [B,H,DH,L]
// MODE 3: f32 out, [M,N]
template <int MODE>
__global__ void gemm64(const __bf16* __restrict__ A, const __bf16* __restrict__ W,
                       void* __restrict__ outp, int M, int N, int K) {
  __shared__ __bf16 lw[64 * 32];           // 4 KB staged weight panel
  const int lane = threadIdx.x & 31;
  const int wave = threadIdx.x >> 5;
  const int ntn  = N >> 6;                 // 64-wide n blocks
  const int tmb  = blockIdx.x / ntn;       // 256-row block index
  const int tn   = blockIdx.x % ntn;
  const int rowb = tmb * 256 + wave * 64;  // this wave's 64-row base
  const int tnb  = tn * 64;

  v8f acc[4][4];
#pragma unroll
  for (int i = 0; i < 4; ++i)
#pragma unroll
    for (int j = 0; j < 4; ++j) acc[i][j] = v8f{0.f,0.f,0.f,0.f,0.f,0.f,0.f,0.f};

  for (int k = 0; k < K; k += 32) {
    __syncthreads();                       // previous panel fully consumed
    // Stage W[tnb..tnb+63, k..k+31] -> LDS (row-major 64x32), 16B per async op.
    {
      const int tid = threadIdx.x;
#pragma unroll
      for (int c0 = 0; c0 < 2; ++c0) {
        const int c    = tid + c0 * 128;   // 256 x 16B chunks
        const int n    = c >> 2;
        const int coff = (c & 3) * 8;
        const __bf16* gp = W + (long)(tnb + n) * K + k + coff;
        const unsigned lo = (unsigned)(size_t)&lw[c * 8];
        asm volatile("global_load_async_to_lds_b128 %0, %1, off"
                     :: "v"(lo), "v"((unsigned long long)(size_t)gp)
                     : "memory");
      }
    }
    // A tiles on the direct VGPR-return path, overlapping the async DMA.
    v16bf a[4];
#pragma unroll
    for (int i = 0; i < 4; ++i) a[i] = load_tile16x32(A, rowb + i * 16, K, k, lane);

    asm volatile("s_wait_asynccnt 0x0" ::: "memory");
    __syncthreads();                       // panel visible to all waves

    v16bf b[4];
#pragma unroll
    for (int j = 0; j < 4; ++j) {
      const int r  = j * 16 + (lane & 15);
      const int kh = (lane >> 4) * 8;
      union { v16bf v; v8bf h[2]; } u;
      u.h[0] = *(const v8bf*)&lw[r * 32 + kh];
      u.h[1] = *(const v8bf*)&lw[r * 32 + kh + 16];
      b[j] = u.v;
    }
#pragma unroll
    for (int i = 0; i < 4; ++i)
#pragma unroll
      for (int j = 0; j < 4; ++j) acc[i][j] = wmma_bf16(a[i], b[j], acc[i][j]);
  }

  const int nloc = lane & 15, half = lane >> 4;
#pragma unroll
  for (int i = 0; i < 4; ++i) {
#pragma unroll
    for (int j = 0; j < 4; ++j) {
      const int n = tnb + j * 16 + nloc;
#pragma unroll
      for (int v = 0; v < 8; ++v) {
        const int m = rowb + i * 16 + half * 8 + v;
        const float val = acc[i][j][v];
        if (MODE == 3) {
          ((float*)outp)[(long)m * N + n] = val;
        } else {
          const int b  = m >> 11, l = m & (LL - 1);
          const int h  = n >> 6,  dh = n & (DH - 1);
          const __bf16 bv = (__bf16)val;
          if (MODE == 0)
            ((__bf16*)outp)[(((long)(b * HH + h) * LL + l) << 6) + dh] = bv;   // [B,H,L,DH]
          else
            ((__bf16*)outp)[(((long)(b * HH + h) * DH + dh) << 11) + l] = bv;  // [B,H,DH,L]
        }
      }
    }
  }
}

// Flash attention: one wave = one (b,h,16-query) tile; Sᵀ tiles feed softmax
// (per-lane + shfl_xor(16) reductions); exp(Sᵀ) packed bf16 is directly the
// B-operand of outᵀ = Vᵀ·Pᵀ — no LDS transpose needed.
__global__ void attn_fa(const __bf16* __restrict__ Qh, const __bf16* __restrict__ Kh,
                        const __bf16* __restrict__ Vt, const float* __restrict__ bias,
                        __bf16* __restrict__ ctx) {
  const int lane = threadIdx.x & 31;
  const int wave = threadIdx.x >> 5;
  const int tile = blockIdx.x * 4 + wave;  // 0..8191
  const int bh = tile >> 7;                // L/16 = 128 query tiles per (b,h)
  const int qt = tile & 127;
  const int b = bh >> 4, h = bh & 15;
  const int q0 = qt << 4;
  const int half = lane >> 4;

  const __bf16* Qb = Qh + (long)bh * LL * DH;
  const __bf16* Kb = Kh + (long)bh * LL * DH;
  const __bf16* Vb = Vt + (long)bh * DH * LL;
  const float*  mb = bias + b * LL;

  const v16bf bq0 = load_tile16x32(Qb, q0, DH, 0,  lane);
  const v16bf bq1 = load_tile16x32(Qb, q0, DH, 32, lane);

  float m_run  = -3.0e38f;
  float l_part = 0.f;
  v8f accO[4];
#pragma unroll
  for (int i = 0; i < 4; ++i) accO[i] = v8f{0.f,0.f,0.f,0.f,0.f,0.f,0.f,0.f};

  for (int s0 = 0; s0 < LL; s0 += 32) {
    float st[16];
#pragma unroll
    for (int sub = 0; sub < 2; ++sub) {
      const int s = s0 + sub * 16;
      const v16bf ak0 = load_tile16x32(Kb, s, DH, 0,  lane);
      const v16bf ak1 = load_tile16x32(Kb, s, DH, 32, lane);
      v8f c = {0.f,0.f,0.f,0.f,0.f,0.f,0.f,0.f};
      c = wmma_bf16(ak0, bq0, c);
      c = wmma_bf16(ak1, bq1, c);
      const float4 b0 = *(const float4*)(mb + s + half * 8);
      const float4 b1 = *(const float4*)(mb + s + half * 8 + 4);
      const float bias8[8] = {b0.x, b0.y, b0.z, b0.w, b1.x, b1.y, b1.z, b1.w};
#pragma unroll
      for (int v = 0; v < 8; ++v) st[sub * 8 + v] = c[v] * SCALE + bias8[v];
    }
    // per-query (column) online softmax
    float tmax = st[0];
#pragma unroll
    for (int i = 1; i < 16; ++i) tmax = fmaxf(tmax, st[i]);
    tmax = fmaxf(tmax, __shfl_xor(tmax, 16, 32));
    const float m_new = fmaxf(m_run, tmax);
    const float resc = __expf(m_run - m_new);
    l_part *= resc;
#pragma unroll
    for (int i = 0; i < 4; ++i)
#pragma unroll
      for (int v = 0; v < 8; ++v) accO[i][v] *= resc;

    union { v16bf v; __bf16 e[16]; } pp;
#pragma unroll
    for (int i = 0; i < 16; ++i) {
      const float p = __expf(st[i] - m_new);
      l_part += p;
      pp.e[i] = (__bf16)p;     // st[0..7]->K=8*half+v, st[8..15]->K=16+8*half+v : exact B layout
    }
    m_run = m_new;
#pragma unroll
    for (int dt = 0; dt < 4; ++dt) {
      const v16bf av = load_tile16x32(Vb, dt * 16, LL, s0, lane);
      accO[dt] = wmma_bf16(av, pp.v, accO[dt]);   // outᵀ[d, query] accumulate
    }
  }

  const float l_tot = l_part + __shfl_xor(l_part, 16, 32);
  const float inv = 1.f / l_tot;
  const int nloc = lane & 15;
  const long l = q0 + nloc;
  __bf16* cb = ctx + ((long)b * LL + l) * DD + h * DH;
#pragma unroll
  for (int dt = 0; dt < 4; ++dt)
#pragma unroll
    for (int v = 0; v < 8; ++v) {
      const int d = dt * 16 + half * 8 + v;
      cb[d] = (__bf16)(accO[dt][v] * inv);
    }
}

extern "C" void kernel_launch(void* const* d_in, const int* in_sizes, int n_in,
                              void* d_out, int out_size, void* d_ws, size_t ws_size,
                              hipStream_t stream) {
  const float* q   = (const float*)d_in[0];
  const float* k   = (const float*)d_in[1];
  const float* v   = (const float*)d_in[2];
  const int*  mask = (const int*)  d_in[3];
  const float* Wq  = (const float*)d_in[4];
  const float* Wk  = (const float*)d_in[5];
  const float* Wv  = (const float*)d_in[6];
  const float* Wo  = (const float*)d_in[7];
  float* out = (float*)d_out;

  const size_t NE = (size_t)BB * LL * DD;   // 8388608
  const size_t WE = (size_t)DD * DD;        // 1048576

  char* ws = (char*)d_ws;
  size_t off = 0;
  auto alloc = [&](size_t bytes) -> void* {
    void* p = ws + off;
    off += (bytes + 255) & ~(size_t)255;
    return p;
  };
  __bf16* qb  = (__bf16*)alloc(NE * 2);
  __bf16* kb  = (__bf16*)alloc(NE * 2);
  __bf16* vb  = (__bf16*)alloc(NE * 2);
  __bf16* wqb = (__bf16*)alloc(WE * 2);
  __bf16* wkb = (__bf16*)alloc(WE * 2);
  __bf16* wvb = (__bf16*)alloc(WE * 2);
  __bf16* wob = (__bf16*)alloc(WE * 2);
  __bf16* Qhp = (__bf16*)alloc(NE * 2);
  __bf16* Khp = (__bf16*)alloc(NE * 2);
  __bf16* Vtp = (__bf16*)alloc(NE * 2);
  float*  bsp = (float*)alloc((size_t)BB * LL * 4);
  __bf16* ctx = qb;  // qb is dead after the Q projection

  const int T = 256;
  cvt_f32_bf16_x8<<<(int)(NE / 8 / T), T, 0, stream>>>((const float4*)q,  (v8bf*)qb,  (int)(NE / 8));
  cvt_f32_bf16_x8<<<(int)(NE / 8 / T), T, 0, stream>>>((const float4*)k,  (v8bf*)kb,  (int)(NE / 8));
  cvt_f32_bf16_x8<<<(int)(NE / 8 / T), T, 0, stream>>>((const float4*)v,  (v8bf*)vb,  (int)(NE / 8));
  cvt_f32_bf16_x8<<<(int)(WE / 8 / T), T, 0, stream>>>((const float4*)Wq, (v8bf*)wqb, (int)(WE / 8));
  cvt_f32_bf16_x8<<<(int)(WE / 8 / T), T, 0, stream>>>((const float4*)Wk, (v8bf*)wkb, (int)(WE / 8));
  cvt_f32_bf16_x8<<<(int)(WE / 8 / T), T, 0, stream>>>((const float4*)Wv, (v8bf*)wvb, (int)(WE / 8));
  cvt_f32_bf16_x8<<<(int)(WE / 8 / T), T, 0, stream>>>((const float4*)Wo, (v8bf*)wob, (int)(WE / 8));
  mask_bias<<<(BB * LL) / T, T, 0, stream>>>(mask, bsp, BB * LL);

  const int M = BB * LL;                               // 8192
  const int gemmBlocks = (M / 256) * (DD / 64);        // 512 blocks, 4 waves each
  gemm64<0><<<gemmBlocks, 128, 0, stream>>>(qb, wqb, Qhp, M, DD, DD);
  gemm64<0><<<gemmBlocks, 128, 0, stream>>>(kb, wkb, Khp, M, DD, DD);
  gemm64<2><<<gemmBlocks, 128, 0, stream>>>(vb, wvb, Vtp, M, DD, DD);

  attn_fa<<<(BB * HH * (LL / 16)) / 4, 128, 0, stream>>>(Qhp, Khp, Vtp, bsp, ctx);

  gemm64<3><<<gemmBlocks, 128, 0, stream>>>(ctx, wob, out, M, DD, DD);
}
